// BaseDGCNNGFModule_69183333204376
// MI455X (gfx1250) — matrix operand order repaired and checked by reading, not compiled, not run on hardware
//
#include <hip/hip_runtime.h>

typedef __attribute__((ext_vector_type(16))) _Float16 v16h;
typedef __attribute__((ext_vector_type(8)))  _Float16 v8h;
typedef __attribute__((ext_vector_type(8)))  float    v8f;

constexpr int BB = 8;     // batch
constexpr int NN = 4096;  // points
constexpr int CC = 64;    // input channels
constexpr int KK = 20;    // neighbors
constexpr int C1 = 128, C2 = 64, C3 = 64;

constexpr int PAD128 = 136;  // padded row stride (halves) for 128-wide rows: 68 dwords -> conflict-free-ish
constexpr int PAD64  = 72;   // padded row stride for 64-wide rows
constexpr int GPW    = 16;   // (b,n) groups per wave in the MLP kernel

// ---------------------------------------------------------------------------
// WMMA helpers (CDNA5 layouts, CDNA5 ISA §7.12.2)
// ---------------------------------------------------------------------------

// A-matrix 16x32 f16 tile from a row-major (stride in halves) buffer.
// lane = M + 16*(K bit3); halves 0-7 = K{0..7}+8*(lane>>4), 8-15 = +16.
__device__ __forceinline__ v16h load_a_tile(const _Float16* base, int stride,
                                            int row0, int k0, int lane) {
  const int m  = row0 + (lane & 15);
  const int kb = k0 + 8 * (lane >> 4);
  const _Float16* p = base + (size_t)m * stride + kb;
  v8h lo = *(const v8h*)p;
  v8h hi = *(const v8h*)(p + 16);
  v16h r;
#pragma unroll
  for (int i = 0; i < 8; ++i) { r[i] = lo[i]; r[i + 8] = hi[i]; }
  return r;
}

// B-matrix 32x16 f16 tile: column N = row N of row-major (N x Kdim) weights.
// lane = N + 16*(K bit3); halves 0-7 = K{0..7}+8*(lane>>4), 8-15 = +16.
__device__ __forceinline__ v16h load_b_tile(const _Float16* base, int stride,
                                            int n0, int k0, int lane) {
  const int nr = n0 + (lane & 15);
  const int kb = k0 + 8 * (lane >> 4);
  const _Float16* p = base + (size_t)nr * stride + kb;
  v8h lo = *(const v8h*)p;
  v8h hi = *(const v8h*)(p + 16);
  v16h r;
#pragma unroll
  for (int i = 0; i < 8; ++i) { r[i] = lo[i]; r[i + 8] = hi[i]; }
  return r;
}

__device__ __forceinline__ v8f wmma16(v16h a, v16h b, v8f c) {
  return __builtin_amdgcn_wmma_f32_16x16x32_f16(false, a, false, b,
                                                (short)0, c, false, false);
}

// gfx1250 async copy: 16B global -> LDS, tracked by ASYNCcnt (ISA §10 / §15.18.3).
// Low 32 bits of a generic shared-space address are the LDS byte offset.
__device__ __forceinline__ void async_copy16(_Float16* lds_dst, const _Float16* gsrc) {
  unsigned lds_off = (unsigned)(size_t)lds_dst;
  asm volatile("global_load_async_to_lds_b128 %0, %1, off"
               :: "v"(lds_off), "v"(gsrc) : "memory");
}

// ---------------------------------------------------------------------------
// Prep: f32 -> f16 points + squared norms; f32 -> f16 weights
// ---------------------------------------------------------------------------
__global__ void prep_points_kernel(const float* __restrict__ pts,
                                   _Float16* __restrict__ pts_h,
                                   float* __restrict__ sq) {
  int i = blockIdx.x * blockDim.x + threadIdx.x;
  if (i >= BB * NN) return;
  const float* p = pts + (size_t)i * CC;
  float s = 0.f;
#pragma unroll 8
  for (int c = 0; c < CC; ++c) {
    float v = p[c];
    s += v * v;
    pts_h[(size_t)i * CC + c] = (_Float16)v;
  }
  sq[i] = s;
}

__global__ void cvt_f16_kernel(const float* __restrict__ src,
                               _Float16* __restrict__ dst, int n) {
  int i = blockIdx.x * blockDim.x + threadIdx.x;
  if (i < n) dst[i] = (_Float16)src[i];
}

// ---------------------------------------------------------------------------
// Distance + top-K: one workgroup = (batch b, 16 rows). Phase 1: 8 waves
// compute Gram tiles via WMMA and scatter d = |a|^2+|b|^2-2ab into LDS
// (16 x 4096 f32 = 256KB, allowed: 320KB/WGP). Phase 2: per-row wave-parallel
// iterative argmin selection of the 20 smallest distances.
// ---------------------------------------------------------------------------
__global__ void dist_topk_kernel(const _Float16* __restrict__ pts_h,
                                 const float* __restrict__ sq,
                                 int* __restrict__ knn_idx) {
  extern __shared__ float distbuf[];  // 16 * NN floats
  const int b    = blockIdx.y;
  const int n0   = blockIdx.x * 16;
  const int lane = threadIdx.x & 31;
  const int wave = threadIdx.x >> 5;

  const _Float16* X   = pts_h + (size_t)b * NN * CC;
  const float*    sqr = sq + (size_t)b * NN;

  // A tiles: rows n0..n0+15, K = 0..63 (two 16x32 tiles)
  v16h a0 = load_a_tile(X, CC, n0, 0, lane);
  v16h a1 = load_a_tile(X, CC, n0, 32, lane);
  float sqa[8];
#pragma unroll
  for (int v = 0; v < 8; ++v) sqa[v] = sqr[n0 + v + 8 * (lane >> 4)];

  for (int j = wave; j < NN / 16; j += 8) {
    // B tile columns = point rows j*16.. (row-major => contiguous K)
    v16h b0 = load_b_tile(X, CC, j * 16, 0, lane);
    v16h b1 = load_b_tile(X, CC, j * 16, 32, lane);
    v8f acc = {};
    acc = wmma16(a0, b0, acc);
    acc = wmma16(a1, b1, acc);
    const int col = j * 16 + (lane & 15);
    const float sqc = sqr[col];
#pragma unroll
    for (int v = 0; v < 8; ++v) {
      int row = v + 8 * (lane >> 4);
      distbuf[row * NN + col] = sqa[v] + sqc - 2.0f * acc[v];
    }
  }
  __syncthreads();

  // Selection: wave w owns rows 2w, 2w+1.
  for (int r = wave * 2; r < wave * 2 + 2; ++r) {
    volatile float* drow = distbuf + (size_t)r * NN;
    int* outp = knn_idx + ((size_t)(b * NN + n0 + r)) * KK;
    for (int kk = 0; kk < KK; ++kk) {
      float best = 3.4e38f;
      int   bi   = 0;
      for (int c = lane; c < NN; c += 32) {
        float dv = drow[c];
        if (dv < best) { best = dv; bi = c; }
      }
#pragma unroll
      for (int off = 16; off > 0; off >>= 1) {
        float ob = __shfl_xor(best, off, 32);
        int   oi = __shfl_xor(bi, off, 32);
        if (ob < best || (ob == best && oi < bi)) { best = ob; bi = oi; }
      }
      if (lane == 0) {
        outp[kk] = bi;
        drow[bi] = 3.4e38f;  // DS ops from this wave stay in order
      }
    }
  }
}

// ---------------------------------------------------------------------------
// Fused 3-layer MLP + max-pool. 8 waves/block, GPW (b,n) groups per wave.
// Weights staged once per workgroup into padded LDS (bank-conflict-free tile
// reads); feat/h buffers per-wave in LDS; all GEMM tiles via
// v_wmma_f32_16x16x32_f16. Max over the 20 neighbors is a column-max of the
// layer-3 D tiles (7 v_max in-register + 1 shfl_xor(16)).
// ---------------------------------------------------------------------------
__global__ void mlp_max_kernel(const _Float16* __restrict__ pts_h,
                               const int* __restrict__ knn_idx,
                               const _Float16* __restrict__ W1h, const float* __restrict__ b1,
                               const _Float16* __restrict__ W2h, const float* __restrict__ b2,
                               const _Float16* __restrict__ W3h, const float* __restrict__ b3,
                               float* __restrict__ out) {
  extern __shared__ _Float16 smem[];
  _Float16* W1L = smem;                      // 128 x PAD128
  _Float16* W2L = W1L + 128 * PAD128;        // 64 x PAD128
  _Float16* W3L = W2L + 64 * PAD128;         // 64 x PAD64
  _Float16* wbuf = W3L + 64 * PAD64;

  const int lane = threadIdx.x & 31;
  const int wave = threadIdx.x >> 5;
  _Float16* feat = wbuf + (size_t)wave * (2 * 32 * PAD128);
  _Float16* hbuf = feat + 32 * PAD128;

  // ---- stage weights into LDS with async global->LDS copies (16B chunks)
  {
    const int tid = threadIdx.x;
    for (int c = tid; c < 128 * 16; c += 256) {       // W1: 128 rows x 16 chunks
      int r = c >> 4, cc = (c & 15) * 8;
      async_copy16(&W1L[r * PAD128 + cc], &W1h[r * 128 + cc]);
    }
    for (int c = tid; c < 64 * 16; c += 256) {        // W2: 64 x 16
      int r = c >> 4, cc = (c & 15) * 8;
      async_copy16(&W2L[r * PAD128 + cc], &W2h[r * 128 + cc]);
    }
    for (int c = tid; c < 64 * 8; c += 256) {         // W3: 64 x 8
      int r = c >> 3, cc = (c & 7) * 8;
      async_copy16(&W3L[r * PAD64 + cc], &W3h[r * 64 + cc]);
    }
    asm volatile("s_wait_asynccnt 0x0" ::: "memory");
  }
  __syncthreads();

  for (int g = 0; g < GPW; ++g) {
    const int bn = (blockIdx.x * 8 + wave) * GPW + g;
    const int b  = bn >> 12;
    const int n  = bn & (NN - 1);

    // ---- build feat rows (lane r owns row r): [edge(64) | center(64)] f16
    {
      const int r = lane;
      const int k = (r < KK) ? r : (KK - 1);
      const int j = knn_idx[(size_t)bn * KK + k];
      const _Float16* pj = pts_h + ((size_t)b * NN + j) * CC;
      const _Float16* pc = pts_h + ((size_t)b * NN + n) * CC;
#pragma unroll
      for (int c0 = 0; c0 < CC; c0 += 8) {
        v8h pv = *(const v8h*)(pj + c0);
        v8h cv = *(const v8h*)(pc + c0);
        v8h ev;
#pragma unroll
        for (int i = 0; i < 8; ++i) ev[i] = pv[i] - cv[i];
        *(v8h*)(feat + r * PAD128 + c0)      = ev;
        *(v8h*)(feat + r * PAD128 + 64 + c0) = cv;
      }
    }
    __syncthreads();

    // ---- layer 1: (32x128) @ W1^T(128x128) -> hbuf (32 x PAD128)
    for (int mt = 0; mt < 2; ++mt)
      for (int nt = 0; nt < 8; ++nt) {
        v8f acc = {};
#pragma unroll
        for (int kt = 0; kt < 4; ++kt) {
          v16h a = load_a_tile(feat, PAD128, mt * 16, kt * 32, lane);
          v16h w = load_b_tile(W1L, PAD128, nt * 16, kt * 32, lane);
          acc = wmma16(a, w, acc);
        }
        const float bias = b1[nt * 16 + (lane & 15)];
#pragma unroll
        for (int v = 0; v < 8; ++v) {
          float x = fmaxf(acc[v] + bias, 0.f);
          hbuf[(mt * 16 + v + 8 * (lane >> 4)) * PAD128 + nt * 16 + (lane & 15)] = (_Float16)x;
        }
      }
    __syncthreads();

    // ---- layer 2: (32x128) @ W2^T(128x64) -> feat region reused (32 x PAD64)
    for (int mt = 0; mt < 2; ++mt)
      for (int nt = 0; nt < 4; ++nt) {
        v8f acc = {};
#pragma unroll
        for (int kt = 0; kt < 4; ++kt) {
          v16h a = load_a_tile(hbuf, PAD128, mt * 16, kt * 32, lane);
          v16h w = load_b_tile(W2L, PAD128, nt * 16, kt * 32, lane);
          acc = wmma16(a, w, acc);
        }
        const float bias = b2[nt * 16 + (lane & 15)];
#pragma unroll
        for (int v = 0; v < 8; ++v) {
          float x = fmaxf(acc[v] + bias, 0.f);
          feat[(mt * 16 + v + 8 * (lane >> 4)) * PAD64 + nt * 16 + (lane & 15)] = (_Float16)x;
        }
      }
    __syncthreads();

    // ---- layer 3 + max over 32 rows (column max of D tiles)
    for (int nt = 0; nt < 4; ++nt) {
      v8f cmax = {};
      for (int mt = 0; mt < 2; ++mt) {
        v8f acc = {};
#pragma unroll
        for (int kt = 0; kt < 2; ++kt) {
          v16h a = load_a_tile(feat, PAD64, mt * 16, kt * 32, lane);
          v16h w = load_b_tile(W3L, PAD64, nt * 16, kt * 32, lane);
          acc = wmma16(a, w, acc);
        }
        const float bias = b3[nt * 16 + (lane & 15)];
#pragma unroll
        for (int v = 0; v < 8; ++v)
          cmax[v] = fmaxf(cmax[v], fmaxf(acc[v] + bias, 0.f));
      }
      float m = cmax[0];
#pragma unroll
      for (int v = 1; v < 8; ++v) m = fmaxf(m, cmax[v]);
      m = fmaxf(m, __shfl_xor(m, 16, 32));  // lanes l and l^16 hold same column N
      if (lane < 16) out[(size_t)bn * C3 + nt * 16 + lane] = m;
    }
    __syncthreads();
  }
}

// ---------------------------------------------------------------------------
extern "C" void kernel_launch(void* const* d_in, const int* in_sizes, int n_in,
                              void* d_out, int out_size, void* d_ws, size_t ws_size,
                              hipStream_t stream) {
  const float* points = (const float*)d_in[0];
  const float* W1 = (const float*)d_in[1];
  const float* b1 = (const float*)d_in[2];
  const float* W2 = (const float*)d_in[3];
  const float* b2 = (const float*)d_in[4];
  const float* W3 = (const float*)d_in[5];
  const float* b3 = (const float*)d_in[6];
  float* out = (float*)d_out;

  char* ws = (char*)d_ws;
  _Float16* pts_h = (_Float16*)ws;  ws += (size_t)BB * NN * CC * 2;   // 4 MB
  float*    sq    = (float*)ws;     ws += (size_t)BB * NN * 4;        // 128 KB
  _Float16* W1h   = (_Float16*)ws;  ws += (size_t)C1 * 2 * CC * 2;    // 32 KB
  _Float16* W2h   = (_Float16*)ws;  ws += (size_t)C2 * C1 * 2;        // 16 KB
  _Float16* W3h   = (_Float16*)ws;  ws += (size_t)C3 * C2 * 2;        // 8 KB
  int*      knn   = (int*)ws;       ws += (size_t)BB * NN * KK * 4;   // 2.6 MB

  prep_points_kernel<<<(BB * NN + 255) / 256, 256, 0, stream>>>(points, pts_h, sq);
  cvt_f16_kernel<<<(C1 * 2 * CC + 255) / 256, 256, 0, stream>>>(W1, W1h, C1 * 2 * CC);
  cvt_f16_kernel<<<(C2 * C1 + 255) / 256, 256, 0, stream>>>(W2, W2h, C2 * C1);
  cvt_f16_kernel<<<(C3 * C2 + 255) / 256, 256, 0, stream>>>(W3, W3h, C3 * C2);

  dim3 gdist(NN / 16, BB);
  dist_topk_kernel<<<gdist, 256, 16 * NN * sizeof(float), stream>>>(pts_h, sq, knn);

  // LDS: W1 (128*136) + W2 (64*136) + W3 (64*72) + 8 waves * 2 * 32*136 halves
  const size_t mlp_lds =
      ((size_t)128 * PAD128 + 64 * PAD128 + 64 * PAD64 + 8 * 2 * 32 * PAD128) *
      sizeof(_Float16);  // ~200 KB
  const int mlp_blocks = (BB * NN) / (8 * GPW);  // 256
  mlp_max_kernel<<<mlp_blocks, 256, mlp_lds, stream>>>(pts_h, knn, W1h, b1,
                                                       W2h, b2, W3h, b3, out);
}
